// AttentionHead_16492674417233
// MI455X (gfx1250) — matrix operand order repaired and checked by reading, not compiled, not run on hardware
//
#include <hip/hip_runtime.h>
#include <hip/hip_bf16.h>

// ---- problem constants (from reference) ----
#define B_DIM 4
#define T_SEQ 4096
#define E_DIM 1024
#define D_DIM 64

typedef __attribute__((ext_vector_type(16))) __bf16 v16bf;
typedef __attribute__((ext_vector_type(8)))  __bf16 v8bf;
typedef __attribute__((ext_vector_type(8)))  float  v8f;

static __device__ inline v8f wmma_bf16(v16bf a, v16bf b, v8f c) {
    // D = A(16x32 bf16) * B(32x16 bf16) + C(16x16 f32)
    return __builtin_amdgcn_wmma_f32_16x16x32_bf16(
        /*neg_a=*/false, a, /*neg_b=*/false, b,
        /*c_mod=*/(short)0, c, /*reuse_a=*/false, /*reuse_b=*/false);
}

// load 16 consecutive bf16 (32 bytes) as one fragment's per-lane data
static __device__ inline v16bf ld16(const __bf16* p) {
    v8bf lo = *(const v8bf*)p;
    v8bf hi = *(const v8bf*)(p + 8);
    return __builtin_shufflevector(lo, hi, 0,1,2,3,4,5,6,7,8,9,10,11,12,13,14,15);
}

static __device__ inline v16bf pack16(float4 a, float4 b, float4 c, float4 d) {
    v16bf r;
    r[0]=(__bf16)a.x;  r[1]=(__bf16)a.y;  r[2]=(__bf16)a.z;  r[3]=(__bf16)a.w;
    r[4]=(__bf16)b.x;  r[5]=(__bf16)b.y;  r[6]=(__bf16)b.z;  r[7]=(__bf16)b.w;
    r[8]=(__bf16)c.x;  r[9]=(__bf16)c.y;  r[10]=(__bf16)c.z; r[11]=(__bf16)c.w;
    r[12]=(__bf16)d.x; r[13]=(__bf16)d.y; r[14]=(__bf16)d.z; r[15]=(__bf16)d.w;
    return r;
}

// ---------------------------------------------------------------------------
// Kernel 1: QKV projection.  One wave per 16(token) x 16(d) tile.
//   grid.x = BT/16 token tiles, grid.y = 12 (3 matrices x 4 d-tiles).
//   Q,K -> row-major bf16 [BT][64];  V -> transposed bf16 [B][64][T].
// ---------------------------------------------------------------------------
__global__ __launch_bounds__(32)
void qkv_kernel(const float* __restrict__ x,
                const float* __restrict__ Wq,
                const float* __restrict__ Wk,
                const float* __restrict__ Wv,
                __bf16* __restrict__ Qb,
                __bf16* __restrict__ Kb,
                __bf16* __restrict__ Vt)
{
    const int lane  = threadIdx.x & 31;
    const int row16 = lane & 15;          // A-row / B-column within tile
    const int half  = lane >> 4;          // K-group select (ISA 7.12.2)
    const int mt    = blockIdx.x;         // token tile
    const int y     = blockIdx.y;
    const int mat   = y >> 2;             // 0=Q 1=K 2=V
    const int nt    = y & 3;              // d tile (16 cols)

    const float* W = (mat == 0) ? Wq : (mat == 1) ? Wk : Wv;
    const int token0 = mt * 16;
    const int dcol   = nt * 16 + row16;

    const float* xrow = x + (size_t)(token0 + row16) * E_DIM;

    v8f acc = {};
    for (int e = 0; e < E_DIM; e += 32) {
        // A operand: x row, bf16-converted.  Per-lane K chunks:
        //   elems 0..7  <- e + 8*half .. +7
        //   elems 8..15 <- e + 16 + 8*half .. +7
        float4 a0 = *(const float4*)(xrow + e + 8 * half);
        float4 a1 = *(const float4*)(xrow + e + 8 * half + 4);
        float4 a2 = *(const float4*)(xrow + e + 16 + 8 * half);
        float4 a3 = *(const float4*)(xrow + e + 16 + 8 * half + 4);
        v16bf af = pack16(a0, a1, a2, a3);

        // B operand: W[e..][dcol]; lane holds column dcol,
        // k = e + 16*(i>>3) + 8*half + (i&7)  (stride D in memory)
        v16bf bfv;
        #pragma unroll
        for (int i = 0; i < 16; ++i)
            bfv[i] = (__bf16)W[(size_t)(e + 16 * (i >> 3) + 8 * half + (i & 7)) * D_DIM + dcol];

        acc = wmma_bf16(af, bfv, acc);
    }

    // C layout: VGPR r -> token row (r + 8*half), col = dcol
    if (mat < 2) {
        __bf16* dst = (mat == 0) ? Qb : Kb;
        #pragma unroll
        for (int r = 0; r < 8; ++r)
            dst[(size_t)(token0 + r + 8 * half) * D_DIM + dcol] = (__bf16)acc[r];
    } else {
        #pragma unroll
        for (int r = 0; r < 8; ++r) {
            int tok = token0 + r + 8 * half;
            int b = tok >> 12;            // / T_SEQ
            int t = tok & (T_SEQ - 1);
            Vt[((size_t)b * D_DIM + dcol) * T_SEQ + t] = (__bf16)acc[r];
        }
    }
}

// ---------------------------------------------------------------------------
// Kernel 2: flash attention. 4 waves/block, each wave owns a 16-query tile
// and streams all 4096 keys in 32-key steps with online softmax.
// ---------------------------------------------------------------------------
__global__ __launch_bounds__(128)
void attn_kernel(const __bf16* __restrict__ Qb,
                 const __bf16* __restrict__ Kb,
                 const __bf16* __restrict__ Vt,
                 float* __restrict__ out)
{
    __shared__ float lds[4][16][36];      // per-wave P staging, padded rows

    const int lane  = threadIdx.x & 31;
    const int wave  = threadIdx.x >> 5;
    const int row16 = lane & 15;
    const int half  = lane >> 4;

    const int qt    = blockIdx.x * 4 + wave;     // 0..1023 query tiles
    const int batch = qt >> 8;                   // 256 tiles / batch
    const int qbase = (qt & 255) << 4;

    // Preload Q A-fragments (row = row16), two 32-wide K chunks over D=64
    const __bf16* qrow = Qb + (size_t)(batch * T_SEQ + qbase + row16) * D_DIM;
    v16bf qA0, qA1;
    {
        v8bf l0 = *(const v8bf*)(qrow + 16 * half);
        v8bf h0 = *(const v8bf*)(qrow + 16 * half + 8);
        qA0 = __builtin_shufflevector(l0, h0, 0,1,2,3,4,5,6,7,8,9,10,11,12,13,14,15);
        v8bf l1 = *(const v8bf*)(qrow + 32 + 16 * half);
        v8bf h1 = *(const v8bf*)(qrow + 32 + 16 * half + 8);
        qA1 = __builtin_shufflevector(l1, h1, 0,1,2,3,4,5,6,7,8,9,10,11,12,13,14,15);
    }

    const __bf16* kbat = Kb + (size_t)batch * T_SEQ * D_DIM;
    const __bf16* vbat = Vt + (size_t)batch * D_DIM * T_SEQ;
    float* plds = &lds[wave][0][0];

    v8f o[4];
    #pragma unroll
    for (int d = 0; d < 4; ++d) o[d] = (v8f){};
    float m_i[8], l_i[8];
    #pragma unroll
    for (int r = 0; r < 8; ++r) { m_i[r] = -1e30f; l_i[r] = 0.0f; }

    for (int kb = 0; kb < T_SEQ; kb += 32) {
        // ---- S = Q * K^T  (two key n-tiles of 16, two d-chunks of 32) ----
        const __bf16* krow0 = kbat + (size_t)(kb + row16) * D_DIM;
        const __bf16* krow1 = kbat + (size_t)(kb + 16 + row16) * D_DIM;
        if (kb + 32 < T_SEQ)
            __builtin_prefetch(krow0 + 32 * D_DIM, 0, 0);

        v16bf kB00 = ld16(krow0 + 16 * half);        // keys kb..+15, d 0..31
        v16bf kB01 = ld16(krow0 + 32 + 16 * half);   // keys kb..+15, d 32..63
        v16bf kB10 = ld16(krow1 + 16 * half);        // keys kb+16..+31, d 0..31
        v16bf kB11 = ld16(krow1 + 32 + 16 * half);

        v8f s0 = {}, s1 = {};
        s0 = wmma_bf16(qA0, kB00, s0);
        s0 = wmma_bf16(qA1, kB01, s0);
        s1 = wmma_bf16(qA0, kB10, s1);
        s1 = wmma_bf16(qA1, kB11, s1);

        // ---- online softmax (rows live across 16-lane halves) ----
        #pragma unroll
        for (int r = 0; r < 8; ++r) {
            float mx = fmaxf(s0[r], s1[r]);
            mx = fmaxf(mx, __shfl_xor(mx, 1, 16));
            mx = fmaxf(mx, __shfl_xor(mx, 2, 16));
            mx = fmaxf(mx, __shfl_xor(mx, 4, 16));
            mx = fmaxf(mx, __shfl_xor(mx, 8, 16));
            float mnew  = fmaxf(m_i[r], mx);
            float alpha = __expf(m_i[r] - mnew);
            m_i[r] = mnew;
            float p0 = __expf(s0[r] - mnew);
            float p1 = __expf(s1[r] - mnew);
            s0[r] = p0; s1[r] = p1;
            float rs = p0 + p1;
            rs += __shfl_xor(rs, 1, 16);
            rs += __shfl_xor(rs, 2, 16);
            rs += __shfl_xor(rs, 4, 16);
            rs += __shfl_xor(rs, 8, 16);
            l_i[r] = l_i[r] * alpha + rs;
            #pragma unroll
            for (int d = 0; d < 4; ++d) o[d][r] *= alpha;
        }

        // ---- C-layout P -> LDS -> A-layout bf16 fragment ----
        #pragma unroll
        for (int r = 0; r < 8; ++r) {
            plds[(r + 8 * half) * 36 + row16]      = s0[r];
            plds[(r + 8 * half) * 36 + 16 + row16] = s1[r];
        }
        asm volatile("s_wait_dscnt 0" ::: "memory");
        const float* prow = plds + row16 * 36;
        float4 f0 = *(const float4*)(prow + 16 * half);
        float4 f1 = *(const float4*)(prow + 16 * half + 4);
        float4 f2 = *(const float4*)(prow + 16 * half + 8);
        float4 f3 = *(const float4*)(prow + 16 * half + 12);
        v16bf pA = pack16(f0, f1, f2, f3);
        asm volatile("" ::: "memory");   // keep reads before next-iter stores

        // ---- O += P * V  (V transposed: lane = d column, contiguous keys) ----
        #pragma unroll
        for (int d = 0; d < 4; ++d) {
            const __bf16* vrow =
                vbat + (size_t)(d * 16 + row16) * T_SEQ + kb + 16 * half;
            v16bf vB = ld16(vrow);
            o[d] = wmma_bf16(pA, vB, o[d]);
        }
    }

    // ---- epilogue: out = O / l / sqrt(D) ----
    const float scale = 0.125f;          // 1/sqrt(64)
    float inv_l[8];
    #pragma unroll
    for (int r = 0; r < 8; ++r) inv_l[r] = scale / l_i[r];

    #pragma unroll
    for (int d = 0; d < 4; ++d) {
        #pragma unroll
        for (int r = 0; r < 8; ++r) {
            int rowq = qbase + r + 8 * half;
            out[((size_t)batch * T_SEQ + rowq) * D_DIM + d * 16 + row16] =
                o[d][r] * inv_l[r];
        }
    }
}

extern "C" void kernel_launch(void* const* d_in, const int* in_sizes, int n_in,
                              void* d_out, int out_size, void* d_ws, size_t ws_size,
                              hipStream_t stream) {
    const float* x  = (const float*)d_in[0];
    const float* Wq = (const float*)d_in[1];
    const float* Wk = (const float*)d_in[2];
    const float* Wv = (const float*)d_in[3];
    float* out = (float*)d_out;

    __bf16* Qb = (__bf16*)d_ws;                                   // [BT][64]
    __bf16* Kb = Qb + (size_t)B_DIM * T_SEQ * D_DIM;              // [BT][64]
    __bf16* Vt = Kb + (size_t)B_DIM * T_SEQ * D_DIM;              // [B][64][T]

    dim3 g1((B_DIM * T_SEQ) / 16, 12);
    qkv_kernel<<<g1, 32, 0, stream>>>(x, Wq, Wk, Wv, Qb, Kb, Vt);

    dim3 g2((B_DIM * T_SEQ) / (16 * 4));
    attn_kernel<<<g2, 128, 0, stream>>>(Qb, Kb, Vt, out);
}
